// ImageTransformer_49435073577607
// MI455X (gfx1250) — matrix operand order repaired and checked by reading, not compiled, not run on hardware
//
#include <hip/hip_runtime.h>

#define NB 2
#define NV 6
#define NC 128
#define NP 200
#define NH 8
#define DH 16
#define NF 256
#define NHW 11264
#define NWF 176
#define NPQ 208
#define NQT 13
#define NSPLIT 8
#define HWT_SPLIT 44      /* (11264/32)/8 */
#define QTB 4             /* q-tiles per block */
#define QGR 4             /* ceil(13/4) q-groups */
#define NEG_INF (-1000000000.0f)
#define LOG_EPS (-15.942385f)

typedef __attribute__((ext_vector_type(16))) _Float16 v16h;
typedef __attribute__((ext_vector_type(2)))  _Float16 v2h;
typedef __attribute__((ext_vector_type(8)))  float    v8f;

// ---------------------------------------------------------------- init
__global__ void init_kernel(float* picked, float* onany) {
  int i = blockIdx.x * blockDim.x + threadIdx.x;
  if (i < NB * NP * NC) picked[i] = 0.f;
  if (i < NB * NP) onany[i] = 0.f;
}

// ---------------------------------------------------------------- pred head (shared by stage 1 & 2)
__device__ void pred_head_row(const float* x, const float* w1, const float* w2, float* out) {
  for (int k = 0; k < 18; k++) out[k] = 0.f;
  for (int j = 0; j < NC; j++) {
    float s = 0.f;
    for (int c = 0; c < NC; c++) s += x[c] * w1[c * NC + j];
    s = s > 0.f ? s : 0.f;
    for (int k = 0; k < 18; k++) out[k] += s * w2[j * 18 + k];
  }
}

__global__ void pred1_kernel(const float* feat, const float* qp, const float* w1, const float* w2,
                             float* center1, float* height1, float* dim1, float* rot1, float* heat1) {
  int i = blockIdx.x * blockDim.x + threadIdx.x;
  if (i >= NB * NP) return;
  int b = i / NP, p = i % NP;
  float x[NC];
  for (int c = 0; c < NC; c++) x[c] = feat[(b * NC + c) * NP + p];
  float o[18];
  pred_head_row(x, w1, w2, o);
  center1[(b * 2 + 0) * NP + p] = o[0] + qp[(b * NP + p) * 2 + 0];
  center1[(b * 2 + 1) * NP + p] = o[1] + qp[(b * NP + p) * 2 + 1];
  height1[b * NP + p] = o[2];
  for (int k = 0; k < 3; k++) dim1[(b * 3 + k) * NP + p] = o[3 + k];
  rot1[(b * 2 + 0) * NP + p] = o[6];
  rot1[(b * 2 + 1) * NP + p] = o[7];
  for (int k = 0; k < 10; k++) heat1[(b * 10 + k) * NP + p] = o[8 + k];
}

// ---------------------------------------------------------------- geometry / projection per (b,v,p)
__global__ void geom_kernel(const float* center1, const float* height1, const float* dim1,
                            const float* rot1, const float* l2i, float* geom) {
  int i = blockIdx.x * blockDim.x + threadIdx.x;
  if (i >= NB * NV * NP) return;
  int b = i / (NV * NP), v = (i / NP) % NV, p = i % NP;
  const float SX[8] = {0.5f, 0.5f, 0.5f, 0.5f, -0.5f, -0.5f, -0.5f, -0.5f};
  const float SY[8] = {0.5f, 0.5f, -0.5f, -0.5f, 0.5f, 0.5f, -0.5f, -0.5f};
  const float SZ[8] = {0.f, 1.f, 1.f, 0.f, 0.f, 1.f, 1.f, 0.f};
  float cxw = center1[(b * 2 + 0) * NP + p], cyw = center1[(b * 2 + 1) * NP + p];
  float px = cxw * 0.6f - 54.f, py = cyw * 0.6f - 54.f;
  float hgt = height1[b * NP + p];
  float z = 8.f / (1.f + expf(-hgt)) - 5.f;
  float dx = expf(dim1[(b * 3 + 0) * NP + p]);
  float dy = expf(dim1[(b * 3 + 1) * NP + p]);
  float dz = expf(dim1[(b * 3 + 2) * NP + p]);
  float r0 = rot1[(b * 2 + 0) * NP + p], r1 = rot1[(b * 2 + 1) * NP + p];
  float yaw = atan2f(r0, r1), cy_ = cosf(yaw), sy_ = sinf(yaw);
  const float* M = l2i + (size_t)(b * NV + v) * 16;
  float t0 = M[0] * px + M[1] * py + M[2] * z + M[3];
  float t1 = M[4] * px + M[5] * py + M[6] * z + M[7];
  float t2 = M[8] * px + M[9] * py + M[10] * z + M[11];
  float zc = t2 > 1e-5f ? t2 : 1e-5f;
  float ccx = t0 / zc, ccy = t1 / zc;
  float on = (ccx > 0.f && ccx < 704.f && ccy > 0.f && ccy < 256.f) ? 1.f : 0.f;
  float qvx = ccx * 0.25f, qvy = ccy * 0.25f;
  float mnx = 1e30f, mxx = -1e30f, mny = 1e30f, mxy = -1e30f;
  for (int c8 = 0; c8 < 8; c8++) {
    float lx = dx * SX[c8], ly = dy * SY[c8], lz = dz * SZ[c8];
    float wx = px + lx * cy_ - ly * sy_;
    float wy = py + lx * sy_ + ly * cy_;
    float wz = z + lz;
    float u0 = M[0] * wx + M[1] * wy + M[2] * wz + M[3];
    float u1 = M[4] * wx + M[5] * wy + M[6] * wz + M[7];
    float u2 = M[8] * wx + M[9] * wy + M[10] * wz + M[11];
    float uz = u2 > 1e-5f ? u2 : 1e-5f;
    float ux = u0 / uz, uy = u1 / uz;
    mnx = fminf(mnx, ux); mxx = fmaxf(mxx, ux);
    mny = fminf(mny, uy); mxy = fmaxf(mxy, uy);
  }
  float ex = (mxx - mnx) * 0.25f, ey = (mxy - mny) * 0.25f;
  float radius = ceilf(sqrtf(ex * ex + ey * ey) * 0.5f);
  float sigma = (radius * 2.f + 1.f) / 6.f;
  float* g = geom + (size_t)i * 6;
  g[0] = on; g[1] = qvx; g[2] = qvy;
  g[3] = truncf(qvx); g[4] = truncf(qvy); g[5] = sigma;
}

// ---------------------------------------------------------------- kpe transposed (C, HW)
__global__ void kpe_kernel(const float* fp, const float* w1, const float* w2, float* kpeT) {
  int hw = blockIdx.x, t = threadIdx.x;
  __shared__ float hid[NC];
  float fx = fp[hw * 2], fy = fp[hw * 2 + 1];
  float h = fx * w1[t] + fy * w1[NC + t];
  hid[t] = h > 0.f ? h : 0.f;
  __syncthreads();
  float s = 0.f;
  for (int c = 0; c < NC; c++) s += hid[c] * w2[c * NC + t];
  kpeT[(size_t)t * NHW + hw] = s;
}

// ---------------------------------------------------------------- qpe per view
__global__ void qpe_kernel(const float* geom, const float* w1, const float* w2, float* qpe, int v) {
  int bp = blockIdx.x, t = threadIdx.x;
  int b = bp / NP, p = bp % NP;
  const float* g = geom + (size_t)((b * NV + v) * NP + p) * 6;
  __shared__ float hid[NC];
  float h = g[1] * w1[t] + g[2] * w1[NC + t];
  hid[t] = h > 0.f ? h : 0.f;
  __syncthreads();
  float s = 0.f;
  for (int c = 0; c < NC; c++) s += hid[c] * w2[c * NC + t];
  qpe[(size_t)bp * NC + t] = s;
}

// ---------------------------------------------------------------- self-attn QKV projection
__global__ void selfqkv_kernel(const float* feat, const float* qpe, const float* wqkv, float* sqkv) {
  int bp = blockIdx.x, t = threadIdx.x;
  int b = bp / NP, p = bp % NP;
  __shared__ float hv[NC];
  hv[t] = feat[(b * NC + t) * NP + p] + qpe[(size_t)bp * NC + t];
  __syncthreads();
  float q = 0.f, k = 0.f, vv2 = 0.f;
  for (int c = 0; c < NC; c++) {
    float h = hv[c];
    const float* w = wqkv + (size_t)c * 3 * NC;
    q += h * w[t]; k += h * w[NC + t]; vv2 += h * w[2 * NC + t];
  }
  float* o = sqkv + (size_t)bp * 3 * NC;
  o[t] = q; o[NC + t] = k; o[2 * NC + t] = vv2;
}

// ---------------------------------------------------------------- self-attention (P keys, masked)
__global__ void selfattn_kernel(const float* sqkv, const float* geom, float* sattn, int v) {
  int i = blockIdx.x * blockDim.x + threadIdx.x;
  if (i >= NB * NH * NP) return;
  int b = i / (NH * NP), h = (i / NP) % NH, p = i % NP;
  const float* qrow = sqkv + (size_t)(b * NP + p) * 3 * NC + h * DH;
  float qv[DH];
#pragma unroll
  for (int d = 0; d < DH; d++) qv[d] = qrow[d];
  const float* gb = geom + (size_t)((b * NV + v) * NP) * 6;
  float m = -1e30f;
  for (int k = 0; k < NP; k++) {
    const float* kr = sqkv + (size_t)(b * NP + k) * 3 * NC + NC + h * DH;
    float s = 0.f;
#pragma unroll
    for (int d = 0; d < DH; d++) s += qv[d] * kr[d];
    s = s * 0.25f + (gb[k * 6] != 0.f ? 0.f : NEG_INF);
    m = fmaxf(m, s);
  }
  float l = 0.f, o[DH];
#pragma unroll
  for (int d = 0; d < DH; d++) o[d] = 0.f;
  for (int k = 0; k < NP; k++) {
    const float* kr = sqkv + (size_t)(b * NP + k) * 3 * NC + NC + h * DH;
    float s = 0.f;
#pragma unroll
    for (int d = 0; d < DH; d++) s += qv[d] * kr[d];
    s = s * 0.25f + (gb[k * 6] != 0.f ? 0.f : NEG_INF);
    float pw = expf(s - m);
    l += pw;
    const float* vr = sqkv + (size_t)(b * NP + k) * 3 * NC + 2 * NC + h * DH;
#pragma unroll
    for (int d = 0; d < DH; d++) o[d] += pw * vr[d];
  }
  float inv = 1.f / l;
  float* od = sattn + (size_t)(b * NP + p) * NC + h * DH;
#pragma unroll
  for (int d = 0; d < DH; d++) od[d] = o[d] * inv;
}

// ---------------------------------------------------------------- self out-proj + LN1 + cross-Q (f16)
__global__ void selfout_kernel(const float* feat, const float* sattn, const float* wout,
                               const float* g, const float* bb, const float* qpe,
                               const float* caqkv, float* x1, _Float16* qf16) {
  int bp = blockIdx.x, t = threadIdx.x;
  int b = bp / NPQ, p = bp % NPQ;
  if (p >= NP) { qf16[(size_t)bp * NC + t] = (_Float16)0.f; return; }
  size_t row = (size_t)b * NP + p;
  __shared__ float sa[NC];
  __shared__ float sr[NC];
  sa[t] = sattn[row * NC + t];
  __syncthreads();
  float o = 0.f;
  for (int c = 0; c < NC; c++) o += sa[c] * wout[c * NC + t];
  float r = feat[(b * NC + t) * NP + p] + o;
  sr[t] = r;
  __syncthreads();
  float mean = 0.f;
  for (int c = 0; c < NC; c++) mean += sr[c];
  mean *= (1.f / NC);
  float var = 0.f;
  for (int c = 0; c < NC; c++) { float d = sr[c] - mean; var += d * d; }
  var *= (1.f / NC);
  float xv = (r - mean) * rsqrtf(var + 1e-5f) * g[t] + bb[t];
  x1[row * NC + t] = xv;
  __syncthreads();
  sa[t] = xv + qpe[row * NC + t];
  __syncthreads();
  float qc = 0.f;
  for (int c = 0; c < NC; c++) qc += sa[c] * caqkv[(size_t)c * 3 * NC + t];
  qf16[(size_t)bp * NC + t] = (_Float16)qc;
}

// ---------------------------------------------------------------- WMMA: Kproj/Vproj = w^T (img+kpe^T)
// kv tile staged once into LDS (packed channel pairs) -> img read exactly once from HBM,
// B fragments are single ds_load_b32 per pair. Kproj stored transposed (B, HW, C).
__global__ void kvproj_kernel(const float* imgfeat, const float* kpeT, const float* caqkv,
                              _Float16* kf16T, _Float16* vf16, int v) {
  int hw0 = blockIdx.x * 16;
  int b = blockIdx.y;
  __shared__ v2h sB2[NC / 2][16];
  const size_t imgbase = (size_t)(b * NV + v) * NC * NHW;
  for (int idx = threadIdx.x; idx < (NC / 2) * 16; idx += 256) {
    int c2 = idx >> 4, hwl = idx & 15;
    int c = 2 * c2;
    float a0 = imgfeat[imgbase + (size_t)c * NHW + hw0 + hwl] +
               kpeT[(size_t)c * NHW + hw0 + hwl];
    float a1 = imgfeat[imgbase + (size_t)(c + 1) * NHW + hw0 + hwl] +
               kpeT[(size_t)(c + 1) * NHW + hw0 + hwl];
    v2h pk;
    pk[0] = (_Float16)a0;
    pk[1] = (_Float16)a1;
    sB2[c2][hwl] = pk;
  }
  __syncthreads();
  int wave = threadIdx.x >> 5, lane = threadIdx.x & 31;
  int half = lane >> 4, mr = lane & 15;
  int m0 = wave * 16;
  v8f accK = {}, accV = {};
#pragma unroll
  for (int ks = 0; ks < 4; ks++) {
    int kb = ks * 32;
    v16h aK, aV, bF;
#pragma unroll
    for (int j = 0; j < 8; j++) {
      int kk = kb + 2 * j + (j >= 4 ? 8 : 0) + 8 * half;
      const float* wA = caqkv + (size_t)kk * 3 * NC;
      const float* wB = caqkv + (size_t)(kk + 1) * 3 * NC;
      aK[2 * j]     = (_Float16)wA[NC + m0 + mr];
      aK[2 * j + 1] = (_Float16)wB[NC + m0 + mr];
      aV[2 * j]     = (_Float16)wA[2 * NC + m0 + mr];
      aV[2 * j + 1] = (_Float16)wB[2 * NC + m0 + mr];
    }
#pragma unroll
    for (int j = 0; j < 8; j++) {
      int p2 = (kb >> 1) + j + (j >= 4 ? 4 : 0) + 4 * half;      // (kb + 2j + 8*(j>=4) + 8*half)/2
      v2h pk = sB2[p2][mr];
      bF[2 * j] = pk[0];
      bF[2 * j + 1] = pk[1];
    }
    accK = __builtin_amdgcn_wmma_f32_16x16x32_f16(false, aK, false, bF, (short)0, accK, false, false);
    accV = __builtin_amdgcn_wmma_f32_16x16x32_f16(false, aV, false, bF, (short)0, accV, false, false);
  }
  // Kproj transposed store: (B, HW, C), channel pairs packed -> b32 stores
#pragma unroll
  for (int r = 0; r < 8; r += 2) {
    int rowo = m0 + r + 8 * half;
    v2h pk;
    pk[0] = (_Float16)accK[r];
    pk[1] = (_Float16)accK[r + 1];
    *(v2h*)(kf16T + ((size_t)b * NHW + hw0 + mr) * NC + rowo) = pk;
  }
  // Vproj in (B, C, HW): contiguous-hw pairs for the V^T A-fragment
#pragma unroll
  for (int r = 0; r < 8; r++) {
    int rowo = m0 + r + 8 * half;
    vf16[((size_t)b * NC + rowo) * NHW + hw0 + mr] = (_Float16)accV[r];
  }
}

// ---------------------------------------------------------------- flash cross-attention (WMMA, 1 wave)
// QTB q-tiles batched per block; K B-fragment is one contiguous 32B v16h load.
__global__ void crossattn_kernel(const _Float16* qf16, const _Float16* kf16T, const _Float16* vf16,
                                 const float* geom, float* part, int v) {
  int qg = blockIdx.x, h = blockIdx.y;
  int b = blockIdx.z / NSPLIT, split = blockIdx.z % NSPLIT;
  int lane = threadIdx.x & 31, half = lane >> 4, n = lane & 15;
  __shared__ float sS[16][32];
  __shared__ v2h sP2[16][16];
  __shared__ float sM[QTB][16], sL[QTB][16], sC[QTB][16];
  __shared__ float gX[QTB * 16], gY[QTB * 16], gI[QTB * 16];
  int qt0 = qg * QTB;
  for (int idx = lane; idx < QTB * 16; idx += 32) {
    int q = qt0 * 16 + idx;
    if (q < NP) {
      const float* g = geom + (size_t)((b * NV + v) * NP + q) * 6;
      gX[idx] = g[3]; gY[idx] = g[4];
      float sg = g[5];
      gI[idx] = 1.f / (2.f * sg * sg);
    } else gI[idx] = -1.f;
  }
  v16h qa[QTB];
  v8f acc[QTB];
  v8f zero = {};
  v16h zh = {};
#pragma unroll
  for (int qi = 0; qi < QTB; qi++) {
    int qt = qt0 + qi;
    acc[qi] = zero;
    qa[qi] = zh;
    if (qt < NQT) {
      const _Float16* qrow = qf16 + (size_t)(b * NPQ + qt * 16 + n) * NC + h * DH;
#pragma unroll
      for (int j = 0; j < 8; j++) {
        if (half == 0) {
          v2h pk = *(const v2h*)(qrow + 2 * j);
          qa[qi][2 * j] = pk[0];
          qa[qi][2 * j + 1] = pk[1];
        }
      }
    }
    if (lane < 16) { sM[qi][lane] = -1e30f; sL[qi][lane] = 0.f; }
  }
  __syncthreads();
  int it0 = split * HWT_SPLIT, it1 = it0 + HWT_SPLIT;
  for (int it = it0; it < it1; it++) {
    int hw0 = it * 32;
    v16h kb0 = zh, kb1 = zh, av;
    if (half == 0) {
      kb0 = *(const v16h*)(kf16T + ((size_t)b * NHW + hw0 + n) * NC + h * DH);
      kb1 = *(const v16h*)(kf16T + ((size_t)b * NHW + hw0 + 16 + n) * NC + h * DH);
    }
    const _Float16* vrb = vf16 + ((size_t)b * NC + h * DH + n) * NHW + hw0;
#pragma unroll
    for (int j = 0; j < 8; j++) {
      int kv2 = 2 * j + (j >= 4 ? 8 : 0) + 8 * half;
      v2h pv = *(const v2h*)(vrb + kv2);
      av[2 * j] = pv[0];
      av[2 * j + 1] = pv[1];
    }
#pragma unroll
    for (int qi = 0; qi < QTB; qi++) {
      int qt = qt0 + qi;
      if (qt >= NQT) break;   // uniform across block
      v8f s0 = zero, s1 = zero;
      s0 = __builtin_amdgcn_wmma_f32_16x16x32_f16(false, qa[qi], false, kb0, (short)0, s0, false, false);
      s1 = __builtin_amdgcn_wmma_f32_16x16x32_f16(false, qa[qi], false, kb1, (short)0, s1, false, false);
#pragma unroll
      for (int r = 0; r < 8; r++) {
        sS[r + 8 * half][n] = s0[r];
        sS[r + 8 * half][16 + n] = s1[r];
      }
      __syncthreads();
      if (lane < 16) {
        int q = qt * 16 + lane;
        float gx = gX[qi * 16 + lane], gy = gY[qi * 16 + lane], gi = gI[qi * 16 + lane];
        float mo = sM[qi][lane], lo = sL[qi][lane];
        float vals[32];
        float mt = mo;
        for (int k = 0; k < 32; k++) {
          float val;
          if (q >= NP) val = NEG_INF;
          else {
            int hwc = hw0 + k;
            float fx = (float)(hwc % NWF), fy = (float)(hwc / NWF);
            float ddx = gx - fx, ddy = gy - fy;
            float lg = -(ddx * ddx + ddy * ddy) * gi;
            float bias = (lg < LOG_EPS) ? NEG_INF : lg;
            val = sS[lane][k] * 0.25f + bias;
          }
          vals[k] = val;
          mt = fmaxf(mt, val);
        }
        float corr = expf(mo - mt), ls = lo * corr;
        for (int k2 = 0; k2 < 16; k2++) {
          float pv0 = expf(vals[2 * k2] - mt);
          float pv1 = expf(vals[2 * k2 + 1] - mt);
          v2h pk;
          pk[0] = (_Float16)pv0;
          pk[1] = (_Float16)pv1;
          sP2[lane][k2] = pk;
          ls += pv0 + pv1;
        }
        sM[qi][lane] = mt; sL[qi][lane] = ls; sC[qi][lane] = corr;
      }
      __syncthreads();
      float f = sC[qi][n];
#pragma unroll
      for (int r = 0; r < 8; r++) acc[qi][r] *= f;
      v16h pb;
#pragma unroll
      for (int j = 0; j < 8; j++) {
        v2h pk = sP2[n][j + (j >= 4 ? 4 : 0) + 4 * half];   // K = 2j + 8*(j>=4) + 8*half
        pb[2 * j] = pk[0];
        pb[2 * j + 1] = pk[1];
      }
      acc[qi] = __builtin_amdgcn_wmma_f32_16x16x32_f16(false, av, false, pb, (short)0, acc[qi], false, false);
      __syncthreads();
    }
  }
#pragma unroll
  for (int qi = 0; qi < QTB; qi++) {
    int qt = qt0 + qi;
    if (qt >= NQT) break;
    float* op = part + ((((size_t)b * NH + h) * NQT + qt) * NSPLIT + split) * 288;
#pragma unroll
    for (int r = 0; r < 8; r++) op[(r + 8 * half) * 16 + n] = acc[qi][r];
    if (lane < 16) { op[256 + lane] = sM[qi][lane]; op[272 + lane] = sL[qi][lane]; }
  }
}

// ---------------------------------------------------------------- merge hw-splits (flash-decoding)
__global__ void combine_kernel(const float* part, float* ocross) {
  int i = blockIdx.x * blockDim.x + threadIdx.x;
  if (i >= NB * NH * NP) return;
  int b = i / (NH * NP), h = (i / NP) % NH, q = i % NP;
  int qt = q / 16, r = q % 16;
  const float* base = part + (((size_t)b * NH + h) * NQT + qt) * NSPLIT * 288;
  float M = -1e30f;
  for (int s = 0; s < NSPLIT; s++) M = fmaxf(M, base[s * 288 + 256 + r]);
  float L = 0.f, w[NSPLIT];
  for (int s = 0; s < NSPLIT; s++) {
    w[s] = expf(base[s * 288 + 256 + r] - M);
    L += base[s * 288 + 272 + r] * w[s];
  }
  float inv = 1.f / L;
  float* od = ocross + (size_t)(b * NP + q) * NC + h * DH;
  for (int d = 0; d < DH; d++) {
    float o = 0.f;
    for (int s = 0; s < NSPLIT; s++) o += base[s * 288 + d * 16 + r] * w[s];
    od[d] = o * inv;
  }
}

// ---------------------------------------------------------------- cross out-proj + LN2 + FFN + LN3 + pick
__global__ void post_kernel(const float* x1, const float* ocross, const float* caout,
                            const float* g2, const float* b2, const float* g3, const float* b3,
                            const float* fw1, const float* fw2, const float* geom,
                            float* picked, float* onany, int v) {
  int bp = blockIdx.x, t = threadIdx.x;
  int b = bp / NP, p = bp % NP;
  size_t row = (size_t)bp;
  __shared__ float s1[NC];
  __shared__ float s2[NC];
  __shared__ float sf[NF];
  s1[t] = ocross[row * NC + t];
  __syncthreads();
  float o = 0.f;
  for (int c = 0; c < NC; c++) o += s1[c] * caout[c * NC + t];
  float r = x1[row * NC + t] + o;
  s2[t] = r;
  __syncthreads();
  float mean = 0.f;
  for (int c = 0; c < NC; c++) mean += s2[c];
  mean *= (1.f / NC);
  float var = 0.f;
  for (int c = 0; c < NC; c++) { float d = s2[c] - mean; var += d * d; }
  var *= (1.f / NC);
  float x2 = (r - mean) * rsqrtf(var + 1e-5f) * g2[t] + b2[t];
  __syncthreads();
  s1[t] = x2;
  __syncthreads();
  for (int jj = 0; jj < 2; jj++) {
    int j = t + jj * NC;
    float hs = 0.f;
    for (int c = 0; c < NC; c++) hs += s1[c] * fw1[c * NF + j];
    sf[j] = hs > 0.f ? hs : 0.f;
  }
  __syncthreads();
  float y = 0.f;
  for (int j = 0; j < NF; j++) y += sf[j] * fw2[j * NC + t];
  float r3 = x2 + y;
  __syncthreads();
  s2[t] = r3;
  __syncthreads();
  float m3 = 0.f;
  for (int c = 0; c < NC; c++) m3 += s2[c];
  m3 *= (1.f / NC);
  float v3 = 0.f;
  for (int c = 0; c < NC; c++) { float d = s2[c] - m3; v3 += d * d; }
  v3 *= (1.f / NC);
  float x3 = (r3 - m3) * rsqrtf(v3 + 1e-5f) * g3[t] + b3[t];
  float on = geom[(size_t)((b * NV + v) * NP + p) * 6];
  if (on != 0.f) {
    picked[row * NC + t] = x3;
    if (t == 0) onany[row] = 1.f;
  }
}

// ---------------------------------------------------------------- final: pred head 2 + outputs
__global__ void final_kernel(const float* picked, const float* onany,
                             const float* w1, const float* w2,
                             const float* center1, const float* height1, const float* dim1,
                             const float* rot1, const float* heat1, float* out) {
  int i = blockIdx.x * blockDim.x + threadIdx.x;
  if (i >= NB * NP) return;
  int b = i / NP, p = i % NP;
  float x[NC];
  const float* pr = picked + (size_t)i * NC;
  for (int c = 0; c < NC; c++) {
    x[c] = pr[c];
    out[(size_t)(b * NC + c) * NP + p] = x[c];   // new_feat (B,C,P)
  }
  float o[18];
  pred_head_row(x, w1, w2, o);
  float on = onany[i];
  float* O1 = out + NB * NC * NP;        // qpos_out (B,P,2)
  float* O2 = O1 + NB * NP * 2;          // center_f (B,2,P)
  float* O3 = O2 + NB * 2 * NP;          // height_f (B,1,P)
  float* O4 = O3 + NB * NP;              // dim_f (B,3,P)
  float* O5 = O4 + NB * 3 * NP;          // rot_f (B,2,P)
  float* O6 = O5 + NB * 2 * NP;          // heatmap_f (B,10,P)
  float* O7 = O6 + NB * 10 * NP;         // on_any (B,P)
  float cf0 = center1[(b * 2 + 0) * NP + p] + (on != 0.f ? o[0] : 0.f);
  float cf1 = center1[(b * 2 + 1) * NP + p] + (on != 0.f ? o[1] : 0.f);
  O1[(b * NP + p) * 2 + 0] = cf0;
  O1[(b * NP + p) * 2 + 1] = cf1;
  O2[(b * 2 + 0) * NP + p] = cf0;
  O2[(b * 2 + 1) * NP + p] = cf1;
  O3[b * NP + p] = on != 0.f ? o[2] : height1[b * NP + p];
  for (int k = 0; k < 3; k++)
    O4[(b * 3 + k) * NP + p] = on != 0.f ? o[3 + k] : dim1[(b * 3 + k) * NP + p];
  O5[(b * 2 + 0) * NP + p] = on != 0.f ? o[6] : rot1[(b * 2 + 0) * NP + p];
  O5[(b * 2 + 1) * NP + p] = on != 0.f ? o[7] : rot1[(b * 2 + 1) * NP + p];
  for (int k = 0; k < 10; k++)
    O6[(b * 10 + k) * NP + p] = on != 0.f ? o[8 + k] : heat1[(b * 10 + k) * NP + p];
  O7[i] = on;
}

// ================================================================ host
extern "C" void kernel_launch(void* const* d_in, const int* in_sizes, int n_in,
                              void* d_out, int out_size, void* d_ws, size_t ws_size,
                              hipStream_t stream) {
  (void)in_sizes; (void)n_in; (void)out_size; (void)ws_size;
  const float* feat   = (const float*)d_in[0];
  const float* qp     = (const float*)d_in[1];
  const float* imgf   = (const float*)d_in[2];
  const float* fpos   = (const float*)d_in[3];
  const float* l2i    = (const float*)d_in[4];
  const float* hw1    = (const float*)d_in[5];
  const float* hw2    = (const float*)d_in[6];
  const float* spw1   = (const float*)d_in[7];
  const float* spw2   = (const float*)d_in[8];
  const float* cpw1   = (const float*)d_in[9];
  const float* cpw2   = (const float*)d_in[10];
  const float* saqkv  = (const float*)d_in[11];
  const float* saout  = (const float*)d_in[12];
  const float* caqkv  = (const float*)d_in[13];
  const float* caout  = (const float*)d_in[14];
  const float* ln1g   = (const float*)d_in[15];
  const float* ln1b   = (const float*)d_in[16];
  const float* ln2g   = (const float*)d_in[17];
  const float* ln2b   = (const float*)d_in[18];
  const float* ln3g   = (const float*)d_in[19];
  const float* ln3b   = (const float*)d_in[20];
  const float* fw1    = (const float*)d_in[21];
  const float* fw2    = (const float*)d_in[22];

  char* ws = (char*)d_ws;
  size_t off = 0;
  auto alloc = [&](size_t bytes) -> char* {
    char* p = ws + off;
    off = (off + bytes + 255) & ~((size_t)255);
    return p;
  };
  float*    center1 = (float*)alloc((size_t)NB * 2 * NP * 4);
  float*    height1 = (float*)alloc((size_t)NB * NP * 4);
  float*    dim1    = (float*)alloc((size_t)NB * 3 * NP * 4);
  float*    rot1    = (float*)alloc((size_t)NB * 2 * NP * 4);
  float*    heat1   = (float*)alloc((size_t)NB * 10 * NP * 4);
  float*    geomb   = (float*)alloc((size_t)NB * NV * NP * 6 * 4);
  float*    kpeT    = (float*)alloc((size_t)NC * NHW * 4);
  float*    qpe     = (float*)alloc((size_t)NB * NP * NC * 4);
  float*    sqkv    = (float*)alloc((size_t)NB * NP * 3 * NC * 4);
  float*    sattn   = (float*)alloc((size_t)NB * NP * NC * 4);
  float*    x1      = (float*)alloc((size_t)NB * NP * NC * 4);
  _Float16* qf16    = (_Float16*)alloc((size_t)NB * NPQ * NC * 2);
  _Float16* kf16T   = (_Float16*)alloc((size_t)NB * NC * NHW * 2);
  _Float16* vf16    = (_Float16*)alloc((size_t)NB * NC * NHW * 2);
  float*    ocross  = (float*)alloc((size_t)NB * NP * NC * 4);
  float*    part    = (float*)alloc((size_t)NB * NH * NQT * NSPLIT * 288 * 4);
  float*    picked  = (float*)alloc((size_t)NB * NP * NC * 4);
  float*    onany   = (float*)alloc((size_t)NB * NP * 4);

  init_kernel<<<(NB * NP * NC + 255) / 256, 256, 0, stream>>>(picked, onany);
  pred1_kernel<<<(NB * NP + 255) / 256, 256, 0, stream>>>(feat, qp, hw1, hw2,
                                                          center1, height1, dim1, rot1, heat1);
  geom_kernel<<<(NB * NV * NP + 255) / 256, 256, 0, stream>>>(center1, height1, dim1, rot1,
                                                              l2i, geomb);
  kpe_kernel<<<NHW, NC, 0, stream>>>(fpos, cpw1, cpw2, kpeT);

  for (int v = 0; v < NV; v++) {
    qpe_kernel<<<NB * NP, NC, 0, stream>>>(geomb, spw1, spw2, qpe, v);
    selfqkv_kernel<<<NB * NP, NC, 0, stream>>>(feat, qpe, saqkv, sqkv);
    selfattn_kernel<<<(NB * NH * NP + 255) / 256, 256, 0, stream>>>(sqkv, geomb, sattn, v);
    selfout_kernel<<<NB * NPQ, NC, 0, stream>>>(feat, sattn, saout, ln1g, ln1b,
                                                qpe, caqkv, x1, qf16);
    kvproj_kernel<<<dim3(NHW / 16, NB), 256, 0, stream>>>(imgf, kpeT, caqkv, kf16T, vf16, v);
    crossattn_kernel<<<dim3(QGR, NH, NB * NSPLIT), 32, 0, stream>>>(qf16, kf16T, vf16,
                                                                    geomb, part, v);
    combine_kernel<<<(NB * NH * NP + 255) / 256, 256, 0, stream>>>(part, ocross);
    post_kernel<<<NB * NP, NC, 0, stream>>>(x1, ocross, caout, ln2g, ln2b, ln3g, ln3b,
                                            fw1, fw2, geomb, picked, onany, v);
  }
  final_kernel<<<(NB * NP + 255) / 256, 256, 0, stream>>>(picked, onany, hw1, hw2,
                                                          center1, height1, dim1, rot1, heat1,
                                                          (float*)d_out);
}
